// DREAMAcousticNL_75677323755738
// MI455X (gfx1250) — compile-verified, matrix operand order checked by reading, not run on hardware
//
#include <hip/hip_runtime.h>
#include <hip/hip_bf16.h>
#include <math.h>

typedef __attribute__((ext_vector_type(16))) _Float16 v16h;
typedef __attribute__((ext_vector_type(8)))  float    v8f;

#define T_LEN 2048
#define NMEL  80
#define NMP   96          // NMEL padded to 3 x 32 for WMMA K
#define HID   512
#define NCLS  64
#define BH    16          // batch rows per half (== WMMA M)
#define NSL   16          // column slices per half
#define SW    32          // slice width (hidden cols per WG)
#define C0R   5           // cell0 input rows per slice (80/16)
#define NWG   32          // 2 halves x 16 slices
#define NT    128         // 4 wave32 per WG
#define DT_C  0.1f

// ---------------- grid-wide barrier (monotonic counter in d_ws) -------------
__device__ inline void grid_sync(unsigned* cnt) {
  __syncthreads();
  if (threadIdx.x == 0) {
    __threadfence();
    unsigned prev = __hip_atomic_fetch_add(cnt, 1u, __ATOMIC_ACQ_REL,
                                           __HIP_MEMORY_SCOPE_AGENT);
    unsigned target = prev - (prev % NWG) + NWG;
    while (__hip_atomic_load(cnt, __ATOMIC_ACQUIRE,
                             __HIP_MEMORY_SCOPE_AGENT) < target) {
      __builtin_amdgcn_s_sleep(8);
    }
  }
  __syncthreads();
}

// ------------- WMMA fragment load: 16-bit 16x32 chunk, row-major [r][k] -----
// ISA layout: lanes 0-15 -> rows 0-15, K pairs {0..7,16..23}; lanes 16-31 ->
// rows 0-15, K pairs {8..15,24..31}.
__device__ inline v16h ldfrag(const _Float16* __restrict__ M0, int ld,
                              int kc, int lane) {
  int r  = lane & 15;
  int kb = (lane & 16) ? 8 : 0;
  const _Float16* p = M0 + r * ld + kc + kb;
  v16h f;
#pragma unroll
  for (int v = 0; v < 4; ++v) { f[2*v] = p[2*v]; f[2*v+1] = p[2*v+1]; }
  p += 16;
#pragma unroll
  for (int v = 0; v < 4; ++v) { f[8+2*v] = p[2*v]; f[8+2*v+1] = p[2*v+1]; }
  return f;
}

// One 16x16 output tile: D = A(16xK) * B(KxN slice at n0)^T, K = 32*KSTEPS.
// B stored as weight rows [n][k] (row-major, ld = K stride).
template<int KSTEPS>
__device__ inline void wmma_tile(const _Float16* __restrict__ A,
                                 const _Float16* __restrict__ Bw,
                                 int ld, float* __restrict__ acc, int accld,
                                 int n0, int lane) {
  v8f c = {};
  const _Float16* Bb = Bw + n0 * ld;
#pragma unroll
  for (int ks = 0; ks < KSTEPS; ++ks) {
    v16h a = ldfrag(A,  ld, ks * 32, lane);
    v16h b = ldfrag(Bb, ld, ks * 32, lane);
    c = __builtin_amdgcn_wmma_f32_16x16x32_f16(false, a, false, b,
                                               (short)0, c, false, false);
  }
  int n  = lane & 15;
  int mb = (lane & 16) ? 8 : 0;
#pragma unroll
  for (int v = 0; v < 8; ++v) acc[(mb + v) * accld + n0 + n] = c[v];
}

// ---------------------------------------------------------------------------
__global__ void __launch_bounds__(NT, 1)
dream_zero(float* __restrict__ out, unsigned* cnt) {
  size_t n = (size_t)32 * T_LEN * NCLS;
  size_t i = (size_t)blockIdx.x * blockDim.x + threadIdx.x;
  if (i < n) out[i] = 0.f;
  if (i == 0) *cnt = 0u;
}

__global__ void __launch_bounds__(NT, 1)
dream_persist(const float* __restrict__ feats,
              const float* __restrict__ C0, const float* __restrict__ B0,
              const float* __restrict__ W0, const float* __restrict__ tau00,
              const float* __restrict__ gam0, const float* __restrict__ ltc0,
              const float* __restrict__ C1, const float* __restrict__ B1,
              const float* __restrict__ W1, const float* __restrict__ tau01,
              const float* __restrict__ gam1, const float* __restrict__ ltc1,
              const float* __restrict__ hw, const float* __restrict__ hb,
              float* __restrict__ out, unsigned* cnt,
              _Float16* __restrict__ H0g, _Float16* __restrict__ H1g,
              float* __restrict__ BE0, float* __restrict__ E0,
              _Float16* __restrict__ E1g, float* __restrict__ PS0,
              float* __restrict__ PSB, float* __restrict__ PS1) {
  const int bh   = blockIdx.x >> 4;   // batch half
  const int sl   = blockIdx.x & 15;   // column slice
  const int col0 = sl * SW;
  const int i0   = sl * C0R;
  const int tid  = threadIdx.x;
  const int lane = tid & 31;
  const int wv   = tid >> 5;

  // ---------------- LDS carve (dynamic, ~198 KB) ----------------
  extern __shared__ char smem[];
  _Float16* C1s  = (_Float16*)smem;         // [SW][HID]
  _Float16* B1s  = C1s + SW * HID;
  _Float16* W1s  = B1s + SW * HID;
  _Float16* C0sp = W1s + SW * HID;          // [16][HID]  rows >= C0R zero
  _Float16* B0s  = C0sp + 16 * HID;         // [SW][NMP]
  _Float16* W0s  = B0s + SW * NMP;
  _Float16* h1f  = W0s + SW * NMP;          // [BH][HID] full h1 (f16)
  _Float16* xn1  = h1f + BH * HID;          // [BH][HID] x_norm1
  _Float16* e1f  = xn1 + BH * HID;          // [BH][HID] full err1; aliased as h0f
  _Float16* xn0  = e1f + BH * HID;          // [BH][NMP]
  _Float16* e0f  = xn0 + BH * NMP;          // [BH][NMP]
  float* x0f  = (float*)(e0f + BH * NMP);   // [BH][NMEL]
  float* accA = x0f + BH * NMEL;            // [BH][SW]; with accB also used as
  float* accB = accA + BH * SW;             //   4x [16][16] partial tiles
  float* h0s  = accB + BH * SW;             // persistent h slices (f32)
  float* h1s  = h0s + BH * SW;
  float* be1s = h1s + BH * SW;
  float* xs0  = be1s + BH * SW;             // [BH]
  float* xs1  = xs0 + BH;
  float* sp0  = xs1 + BH;
  float* sp1  = sp0 + BH;
  float* te0  = sp1 + BH;                   // [SW]  DT / tau_eff
  float* te1  = te0 + SW;
  float* psA  = te1 + SW;                   // [BH] per-row partial sums
  _Float16* h0f  = e1f;                     // alias (disjoint live ranges)
  float*    pacc = accA;                    // alias: 4 x [16][16] partials

  // ---------------- setup: convert weight slices to f16 LDS ----------------
  for (int idx = tid; idx < SW * HID; idx += NT) {
    int r = idx / HID, k = idx % HID;
    C1s[idx] = (_Float16)C1[(col0 + r) * HID + k];
    B1s[idx] = (_Float16)B1[(col0 + r) * HID + k];
    W1s[idx] = (_Float16)W1[(col0 + r) * HID + k];
  }
  for (int idx = tid; idx < 16 * HID; idx += NT) {
    int r = idx / HID, k = idx % HID;
    C0sp[idx] = (r < C0R) ? (_Float16)C0[(i0 + r) * HID + k] : (_Float16)0.f;
  }
  for (int idx = tid; idx < SW * NMP; idx += NT) {
    int r = idx / NMP, k = idx % NMP;
    float b0 = (k < NMEL) ? B0[(col0 + r) * NMEL + k] : 0.f;
    float w0 = (k < NMEL) ? W0[(col0 + r) * NMEL + k] : 0.f;
    B0s[idx] = (_Float16)b0;
    W0s[idx] = (_Float16)w0;
  }
  if (tid < SW) {
    te0[tid] = DT_C / (1.f + log1pf(expf(ltc0[col0 + tid])));
    te1[tid] = DT_C / (1.f + log1pf(expf(ltc1[col0 + tid])));
  }
  for (int idx = tid; idx < BH * SW; idx += NT) {
    h0s[idx] = 0.f; h1s[idx] = 0.f;
    int m = idx / SW, c = idx % SW;
    H0g[(bh * BH + m) * HID + col0 + c] = (_Float16)0.f;
    H1g[(bh * BH + m) * HID + col0 + c] = (_Float16)0.f;
  }
  const float ft00 = tau00[0], fg0 = gam0[0];
  const float ft01 = tau01[0], fg1 = gam1[0];
  grid_sync(cnt);

  // ---------------- time loop ----------------
  for (int t = 0; t < T_LEN; ++t) {
    // phase 1: load x_t, x_scale, x_norm0
    if (tid < BH) {
      const float* xr = feats + ((size_t)(bh * BH + tid) * T_LEN + t) * NMEL;
      float ss = 0.f;
      for (int i = 0; i < NMEL; ++i) { float v = xr[i]; x0f[tid*NMEL+i] = v; ss += v*v; }
      float sc = sqrtf(ss); if (sc < 1e-6f) sc = 1e-6f;
      xs0[tid] = sc;
      psA[tid] = 0.f;
      if (t + 1 < T_LEN) __builtin_prefetch(xr + NMEL, 0, 1);  // global_prefetch_b8
    }
    __syncthreads();
    for (int idx = tid; idx < BH * NMP; idx += NT) {
      int m = idx / NMP, i = idx % NMP;
      float v = 0.f;
      if (i < NMEL) {
        v = x0f[m * NMEL + i] / xs0[m];
        v = fminf(1.f, fmaxf(-1.f, v));
      }
      xn0[idx] = (_Float16)v;
    }
    // phase 2: err0 slice via WMMA: tanh(h0 . C0sp^T), K split over 4 waves
    for (int idx = tid; idx < BH * HID; idx += NT)
      h0f[idx] = H0g[bh * BH * HID + idx];
    __syncthreads();
    wmma_tile<4>(h0f + wv * 128, C0sp + wv * 128, HID,
                 pacc + wv * 256, 16, 0, lane);
    __syncthreads();
    if (tid < BH * C0R) {
      int m = tid / C0R, r = tid % C0R;
      float dot = pacc[m * 16 + r]       + pacc[256 + m * 16 + r] +
                  pacc[512 + m * 16 + r] + pacc[768 + m * 16 + r];
      float e = x0f[m * NMEL + i0 + r] - tanhf(dot) * xs0[m];
      E0[(bh * BH + m) * NMEL + i0 + r] = e;
      atomicAdd(&psA[m], e * e);
    }
    __syncthreads();
    if (tid < BH) PS0[(bh * NSL + sl) * BH + tid] = psA[tid];
    __threadfence();
    grid_sync(cnt);  // barrier A

    // phase 3/4: surprise0, base/err_eff0 via WMMA, update h0 slice
    if (tid < BH) {
      float s = 0.f;
      for (int j = 0; j < NSL; ++j) s += PS0[(bh * NSL + j) * BH + tid];
      float rel = fminf(sqrtf(s) / xs0[tid], 4.f);
      sp0[tid] = 1.f / (1.f + expf(-(rel - ft00) / fg0));
      psA[tid] = 0.f;
    }
    for (int idx = tid; idx < BH * NMP; idx += NT) {
      int m = idx / NMP, i = idx % NMP;
      float e = (i < NMEL) ? E0[(bh * BH + m) * NMEL + i] : 0.f;
      e0f[idx] = (_Float16)e;
    }
    __syncthreads();
    {
      int n0 = (wv & 1) * 16;
      if (wv < 2) wmma_tile<3>(xn0, B0s, NMP, accA, SW, n0, lane);  // base_eff0
      else        wmma_tile<3>(e0f, W0s, NMP, accB, SW, n0, lane);  // err_eff0
    }
    __syncthreads();
    for (int idx = tid; idx < BH * SW; idx += NT) {
      int m = idx / SW, c = idx % SW;
      float be = accA[idx], ee = accB[idx], s = sp0[m], h = h0s[idx];
      float ih = h * 0.2f + be * 0.6f + ee * s * 0.2f;
      float alpha = te0[c] * (1.f + s);
      float hn = h + alpha * (tanhf(ih) - h);
      h0s[idx] = hn;
      H0g[(bh * BH + m) * HID + col0 + c] = (_Float16)hn;
      BE0[(bh * BH + m) * HID + col0 + c] = be;
      atomicAdd(&psA[m], be * be);
    }
    __syncthreads();
    if (tid < BH) PSB[(bh * NSL + sl) * BH + tid] = psA[tid];
    __threadfence();
    grid_sync(cnt);  // barrier B

    // phase 5/6: xscale1/x_norm1, err1 slice = x1 - tanh(h1 . C1s)*xscale1
    if (tid < BH) {
      float s = 0.f;
      for (int j = 0; j < NSL; ++j) s += PSB[(bh * NSL + j) * BH + tid];
      float sc = sqrtf(s); if (sc < 1e-6f) sc = 1e-6f;
      xs1[tid] = sc;
      psA[tid] = 0.f;
    }
    __syncthreads();
    for (int idx = tid; idx < BH * HID; idx += NT) {
      int m = idx / HID;
      h1f[idx] = H1g[bh * BH * HID + idx];
      float v = BE0[bh * BH * HID + idx] / xs1[m];
      v = fminf(1.f, fmaxf(-1.f, v));
      xn1[idx] = (_Float16)v;
    }
    __syncthreads();
    if (wv < 2) wmma_tile<16>(h1f, C1s, HID, accA, SW, (wv & 1) * 16, lane);
    __syncthreads();
    for (int idx = tid; idx < BH * SW; idx += NT) {
      int m = idx / SW, c = idx % SW;
      float x1 = BE0[(bh * BH + m) * HID + col0 + c];
      float e  = x1 - tanhf(accA[idx]) * xs1[m];
      E1g[(bh * BH + m) * HID + col0 + c] = (_Float16)e;
      atomicAdd(&psA[m], e * e);
    }
    __syncthreads();
    if (tid < BH) PS1[(bh * NSL + sl) * BH + tid] = psA[tid];
    __threadfence();
    grid_sync(cnt);  // barrier C

    // phase 7/8: surprise1, base/err_eff1 via WMMA, update h1 slice, head
    if (tid < BH) {
      float s = 0.f;
      for (int j = 0; j < NSL; ++j) s += PS1[(bh * NSL + j) * BH + tid];
      float rel = fminf(sqrtf(s) / xs1[tid], 4.f);
      sp1[tid] = 1.f / (1.f + expf(-(rel - ft01) / fg1));
    }
    for (int idx = tid; idx < BH * HID; idx += NT)
      e1f[idx] = E1g[bh * BH * HID + idx];
    __syncthreads();
    {
      int n0 = (wv & 1) * 16;
      if (wv < 2) wmma_tile<16>(xn1, B1s, HID, accA, SW, n0, lane);  // base_eff1
      else        wmma_tile<16>(e1f, W1s, HID, accB, SW, n0, lane);  // err_eff1
    }
    __syncthreads();
    for (int idx = tid; idx < BH * SW; idx += NT) {
      int m = idx / SW, c = idx % SW;
      float be = accA[idx], ee = accB[idx], s = sp1[m], h = h1s[idx];
      float ih = h * 0.2f + be * 0.6f + ee * s * 0.2f;
      float alpha = te1[c] * (1.f + s);
      float hn = h + alpha * (tanhf(ih) - h);
      h1s[idx]  = hn;
      be1s[idx] = be;
      H1g[(bh * BH + m) * HID + col0 + c] = (_Float16)hn;
    }
    __syncthreads();
    // head partial over this k-slice: out += h1s . hw[:, col0] + be1s . hw[:, 512+col0]
    for (int task = tid; task < BH * NCLS; task += NT) {
      int m = task / NCLS, o = task % NCLS;
      const float* wr = hw + o * (2 * HID) + col0;
      float acc = 0.f;
      for (int c = 0; c < SW; ++c)
        acc += h1s[m * SW + c] * wr[c] + be1s[m * SW + c] * wr[HID + c];
      if (sl == 0) acc += hb[o];
      unsafeAtomicAdd(&out[((size_t)(bh * BH + m) * T_LEN + t) * NCLS + o], acc);
    }
    // no barrier needed here (next step's first shared read is covered by A/B)
  }
}

// ---------------------------------------------------------------------------
extern "C" void kernel_launch(void* const* d_in, const int* in_sizes, int n_in,
                              void* d_out, int out_size, void* d_ws,
                              size_t ws_size, hipStream_t stream) {
  const float* feats = (const float*)d_in[0];
  const float* C0    = (const float*)d_in[1];
  const float* B0    = (const float*)d_in[2];
  const float* W0    = (const float*)d_in[3];
  const float* t00   = (const float*)d_in[4];
  const float* g0    = (const float*)d_in[5];
  const float* l0    = (const float*)d_in[6];
  const float* C1    = (const float*)d_in[7];
  const float* B1    = (const float*)d_in[8];
  const float* W1    = (const float*)d_in[9];
  const float* t01   = (const float*)d_in[10];
  const float* g1    = (const float*)d_in[11];
  const float* l1    = (const float*)d_in[12];
  const float* hw    = (const float*)d_in[13];
  const float* hb    = (const float*)d_in[14];
  float* out = (float*)d_out;

  char* ws = (char*)d_ws;
  unsigned* cnt  = (unsigned*)(ws + 0);
  _Float16* H0g  = (_Float16*)(ws + 256);                     // 2*16*512 f16
  _Float16* H1g  = (_Float16*)(ws + 256 + 32768);
  float*    BE0v = (float*)   (ws + 256 + 65536);             // 2*16*512 f32
  float*    E0v  = (float*)   (ws + 256 + 131072);            // 2*16*80 f32
  _Float16* E1g  = (_Float16*)(ws + 256 + 131072 + 12288);    // 2*16*512 f16
  float*    PS0  = (float*)   (ws + 256 + 131072 + 12288 + 32768);
  float*    PSB  = PS0 + 2 * NSL * BH;
  float*    PS1  = PSB + 2 * NSL * BH;

  size_t n_out = (size_t)32 * T_LEN * NCLS;
  dream_zero<<<(unsigned)((n_out + NT - 1) / NT), NT, 0, stream>>>(out, cnt);

  // dynamic LDS bytes (mirrors the in-kernel carve)
  size_t f16elems = (size_t)SW*HID*3 + (size_t)16*HID + (size_t)SW*NMP*2 +
                    (size_t)BH*HID*3 + (size_t)BH*NMP*2;
  size_t f32elems = (size_t)BH*NMEL + (size_t)BH*SW*5 + BH*4 + SW*2 + BH;
  size_t shmem = f16elems * 2 + f32elems * 4;

  dream_persist<<<NWG, NT, shmem, stream>>>(
      feats, C0, B0, W0, t00, g0, l0, C1, B1, W1, t01, g1, l1, hw, hb,
      out, cnt, H0g, H1g, BE0v, E0v, E1g, PS0, PSB, PS1);
}
